// LearningPathOptimizer_33792802685455
// MI455X (gfx1250) — matrix operand order repaired and checked by reading, not compiled, not run on hardware
//
#include <hip/hip_runtime.h>
#include <hip/hip_bf16.h>
#include <stdint.h>

#define B32     32
#define NCON    8192
#define DC      256
#define HSZ     512
#define NH      8
#define PATHLEN 20

typedef __attribute__((ext_vector_type(16))) __bf16 v16bf;
typedef __attribute__((ext_vector_type(8)))  __bf16 v8bf;
typedef __attribute__((ext_vector_type(8)))  float  v8f;

static __device__ __forceinline__ unsigned short f2bfu(float f) {
  unsigned u = __builtin_bit_cast(unsigned, f);
  unsigned r = u + 0x7FFFu + ((u >> 16) & 1u);   // round-to-nearest-even
  return (unsigned short)(r >> 16);
}
static __device__ __forceinline__ __bf16 f2bf(float f) {
  unsigned short s = f2bfu(f);
  return __builtin_bit_cast(__bf16, s);
}
static __device__ __forceinline__ float sigf(float x) { return 1.0f / (1.0f + __expf(-x)); }

// ---------------- one-time conversion / packing ----------------

// ce f32 -> bf16, 4 elems/thread
__global__ void lpo_cvt_ce(const float* __restrict__ src, __bf16* __restrict__ dst, long long n4) {
  long long i = (long long)blockIdx.x * blockDim.x + threadIdx.x;
  if (i >= n4) return;
  float4 v = ((const float4*)src)[i];
  unsigned long long p = (unsigned long long)f2bfu(v.x)
                       | ((unsigned long long)f2bfu(v.y) << 16)
                       | ((unsigned long long)f2bfu(v.z) << 32)
                       | ((unsigned long long)f2bfu(v.w) << 48);
  ((unsigned long long*)dst)[i] = p;
}

// Pack Wcat = [wih; whh] (logical [1024,2048], row-major [k][n]) into WMMA-B register
// layout tiles: tile (nt,kt), lane l: col n = nt*16 + (l&15), K half = (l<16 ? 0..15 : 16..31).
__global__ void lpo_pack_w(const float* __restrict__ wih, const float* __restrict__ whh,
                           __bf16* __restrict__ wp) {
  int gid  = blockIdx.x * blockDim.x + threadIdx.x;   // 128 nt * 32 kt * 32 lanes
  int lane = gid & 31;
  int kt   = (gid >> 5) & 31;
  int n    = (gid >> 10) * 16 + (lane & 15);
  int kb   = kt * 32 + ((lane < 16) ? 0 : 16);
  __bf16* out = wp + (size_t)gid * 16;
#pragma unroll
  for (int v = 0; v < 16; ++v) {
    int k = kb + v;
    float val = (k < HSZ) ? wih[(size_t)k * 2048 + n] : whh[(size_t)(k - HSZ) * 2048 + n];
    out[v] = f2bf(val);
  }
}

__global__ void lpo_bias_add(const float* __restrict__ a, const float* __restrict__ b,
                             float* __restrict__ o, int n) {
  int i = blockIdx.x * blockDim.x + threadIdx.x;
  if (i < n) o[i] = a[i] + b[i];
}

// ---------------- state encoder (tiny) ----------------

__global__ void lpo_enc1(const float* __restrict__ cs, const float* __restrict__ w1,
                         const float* __restrict__ b1, float* __restrict__ hid) {
  int b = blockIdx.x, j = threadIdx.x;            // 32 x 512
  float acc = b1[j];
  const float* x = cs + (size_t)b * 434;
  for (int d = 0; d < 434; ++d) acc += x[d] * w1[(size_t)d * 512 + j];
  hid[(size_t)b * 512 + j] = fmaxf(acc, 0.0f);
}

__global__ void lpo_enc2(const float* __restrict__ hid, const float* __restrict__ w2,
                         const float* __restrict__ b2, float* __restrict__ sf) {
  int b = blockIdx.x, j = threadIdx.x;            // 32 x 256
  float acc = b2[j];
  const float* x = hid + (size_t)b * 512;
  for (int d = 0; d < 512; ++d) acc += x[d] * w2[(size_t)d * 256 + j];
  sf[(size_t)b * 256 + j] = acc;
}

// ---------------- degenerate argmax: first masked index ----------------

__global__ void lpo_idx_path(const int* __restrict__ mask, int* __restrict__ idx0,
                             float* __restrict__ out_path) {
  int b = threadIdx.x;                            // 32 threads
  int idx = 0;
  for (int n = 0; n < NCON; ++n) {
    if (mask[(size_t)b * NCON + n] != 0) { idx = n; break; }
  }
  idx0[b] = idx;
  for (int t = 0; t < PATHLEN; ++t) out_path[b * PATHLEN + t] = (float)idx;
}

__global__ void lpo_set_sel(const float* __restrict__ ce, const int* __restrict__ idx0,
                            float* __restrict__ sel) {
  int b = blockIdx.x, c = threadIdx.x;            // 32 x 256
  sel[(size_t)b * DC + c] = ce[((size_t)b * NCON + idx0[b]) * DC + c];
}

// ---------------- LSTM gates GEMM: G[32,2048] = Xcat[32,1024] @ Wcat ----------------

// Pack Xcat (concat of up to three f32 row-major sources) into WMMA-A register layout:
// tile (mt,kt), lane l: row = mt*16 + (l&15), K 0..15 (lanes<16) or
// 16..31 (lanes>=16). 2*32*32 = 2048 threads; the branchy gather happens ONCE
// here instead of per N-tile in the GEMM.
__global__ void lpo_pack_x(const float* __restrict__ xa, int la,
                           const float* __restrict__ xb, int lb,
                           const float* __restrict__ xc,
                           __bf16* __restrict__ xp) {
  int gid  = blockIdx.x * blockDim.x + threadIdx.x;   // 2 mt * 32 kt * 32 lanes
  int lane = gid & 31;
  int kt   = (gid >> 5) & 31;
  int row  = (gid >> 10) * 16 + (lane & 15);
  int kb   = kt * 32 + ((lane < 16) ? 0 : 16);
  __bf16* out = xp + (size_t)gid * 16;
#pragma unroll
  for (int i = 0; i < 16; ++i) {
    int k = kb + i;
    float x;
    if (k < la) x = xa[(size_t)row * la + k];
    else {
      int k2 = k - la;
      if (k2 < lb) x = xb[(size_t)row * lb + k2];
      else         x = xc[(size_t)row * HSZ + (k2 - lb)];
    }
    out[i] = f2bf(x);
  }
}

// One wave per 16x16 output tile; both operands pre-packed -> pure b128-load + WMMA loop.
__global__ void lpo_gemm_cat(const __bf16* __restrict__ wp, const __bf16* __restrict__ xp,
                             float* __restrict__ G) {
  int l  = threadIdx.x;
  int nt = blockIdx.x & 127;
  int mt = blockIdx.x >> 7;
  const __bf16* ap = xp + ((size_t)mt * 32 * 32) * 16 + (size_t)l * 16;
  const __bf16* bp = wp + ((size_t)nt * 32 * 32) * 16 + (size_t)l * 16;
  v8f acc = {};
#pragma unroll 8
  for (int kt = 0; kt < 32; ++kt) {
    v16bf a  = *(const v16bf*)(ap + (size_t)kt * 512);
    v16bf bm = *(const v16bf*)(bp + (size_t)kt * 512);
    acc = __builtin_amdgcn_wmma_f32_16x16x32_bf16(false, a, false, bm, (short)0, acc, false, false);
  }
  int n = nt * 16 + (l & 15);
  int moff = (l < 16) ? 0 : 8;
#pragma unroll
  for (int r = 0; r < 8; ++r) {
    int b = mt * 16 + r + moff;
    G[(size_t)b * 2048 + n] = acc[r];
  }
}

__global__ void lpo_lstm_update(const float* __restrict__ G, const float* __restrict__ bias,
                                float* __restrict__ h, float* __restrict__ c) {
  int b = blockIdx.x, j = threadIdx.x;            // 32 x 512
  const float* g = G + (size_t)b * 2048;
  float i_ = g[j]        + bias[j];
  float f_ = g[512 + j]  + bias[512 + j];
  float gg = g[1024 + j] + bias[1024 + j];
  float o_ = g[1536 + j] + bias[1536 + j];
  float cp = c[(size_t)b * 512 + j];
  float c2 = sigf(f_) * cp + sigf(i_) * tanhf(gg);
  c[(size_t)b * 512 + j] = c2;
  h[(size_t)b * 512 + j] = sigf(o_) * tanhf(c2);
}

// ---------------- final-step attention weights ----------------

__global__ void lpo_qproj(const float* __restrict__ h1, const float* __restrict__ wq,
                          const float* __restrict__ bq, float* __restrict__ q) {
  int b = blockIdx.x, j = threadIdx.x;            // 32 x 512
  float acc = bq[j];
  const float* x = h1 + (size_t)b * 512;
  for (int d = 0; d < 512; ++d) acc += x[d] * wq[(size_t)d * 512 + j];
  q[(size_t)b * 512 + j] = acc;
}

// qtil[b,h,c] = (1/8) * sum_d q[b,64h+d] * wk[c,64h+d], packed into WMMA-B tiles
// (col = head padded to 16, K = c in 32-chunks).
__global__ void lpo_qtil(const float* __restrict__ q, const float* __restrict__ wk,
                         __bf16* __restrict__ qtilB) {
  int gid = blockIdx.x * blockDim.x + threadIdx.x;  // 32 * 16 * 256
  int c   = gid & 255;
  int h16 = (gid >> 8) & 15;
  int b   = gid >> 12;
  float val = 0.0f;
  if (h16 < 8) {
    const float* qp = q  + (size_t)b * 512 + h16 * 64;
    const float* wp = wk + (size_t)c * 512 + h16 * 64;
#pragma unroll 8
    for (int d = 0; d < 64; ++d) val += qp[d] * wp[d];
    val *= 0.125f;                                  // 1/sqrt(64)
  }
  int kt = c >> 5, kn = c & 31;
  int lane = h16 + ((kn < 16) ? 0 : 16);
  qtilB[(((size_t)b * 8 + kt) * 32 + lane) * 16 + (kn & 15)] = f2bf(val);
}

// logits[b,h,n] = ce_bf16[b,n,:] . qtil[b,h,:]  via WMMA, one wave per (b, 16-row n-tile)
__global__ void lpo_logits(const __bf16* __restrict__ ceb, const __bf16* __restrict__ qtilB,
                           float* __restrict__ Lg) {
  int l  = threadIdx.x;
  int mt = blockIdx.x & 511;
  int b  = blockIdx.x >> 9;
  const __bf16* arow = ceb + ((size_t)b * NCON + mt * 16 + (l & 15)) * DC + ((l < 16) ? 0 : 16);
  const __bf16* bp   = qtilB + (size_t)b * 8 * 32 * 16;
  v8f acc = {};
#pragma unroll
  for (int kt = 0; kt < 8; ++kt) {
    v8bf lo = *(const v8bf*)(arow + kt * 32);
    v8bf hi = *(const v8bf*)(arow + kt * 32 + 8);
    v16bf a;
#pragma unroll
    for (int i = 0; i < 8; ++i) { a[i] = lo[i]; a[i + 8] = hi[i]; }
    v16bf bm = *(const v16bf*)(bp + ((size_t)kt * 32 + l) * 16);
    acc = __builtin_amdgcn_wmma_f32_16x16x32_bf16(false, a, false, bm, (short)0, acc, false, false);
  }
  int h = l & 15;
  if (h < 8) {
    int moff = (l < 16) ? 0 : 8;
#pragma unroll
    for (int r = 0; r < 8; ++r) {
      int n = mt * 16 + r + moff;
      Lg[((size_t)b * NH + h) * NCON + n] = acc[r];
    }
  }
}

__global__ void lpo_stats(const float* __restrict__ Lg, float* __restrict__ stats) {
  __shared__ float sm[256];
  int bh = blockIdx.x, t = threadIdx.x;           // 256 blocks x 256 threads
  const float* p = Lg + (size_t)bh * NCON;
  float m = -3.0e38f;
  for (int n = t; n < NCON; n += 256) m = fmaxf(m, p[n]);
  sm[t] = m; __syncthreads();
  for (int s = 128; s > 0; s >>= 1) { if (t < s) sm[t] = fmaxf(sm[t], sm[t + s]); __syncthreads(); }
  m = sm[0]; __syncthreads();
  float sum = 0.0f;
  for (int n = t; n < NCON; n += 256) sum += __expf(p[n] - m);
  sm[t] = sum; __syncthreads();
  for (int s = 128; s > 0; s >>= 1) { if (t < s) sm[t] += sm[t + s]; __syncthreads(); }
  if (t == 0) { stats[bh * 2] = m; stats[bh * 2 + 1] = sm[0]; }
}

__global__ void lpo_avg(const float* __restrict__ Lg, const float* __restrict__ stats,
                        float* __restrict__ out) {
  int gid = blockIdx.x * blockDim.x + threadIdx.x;  // 32*8192
  int n = gid & (NCON - 1);
  int b = gid >> 13;
  float s = 0.0f;
#pragma unroll
  for (int h = 0; h < 8; ++h) {
    int bh = b * 8 + h;
    s += __expf(Lg[(size_t)bh * NCON + n] - stats[bh * 2]) / stats[bh * 2 + 1];
  }
  out[gid] = 0.125f * s;
}

// ---------------- quality head ----------------

__global__ void lpo_qe1(const float* __restrict__ h1, const float* __restrict__ w1,
                        const float* __restrict__ b1, float* __restrict__ hq) {
  int b = blockIdx.x, j = threadIdx.x;            // 32 x 256
  float acc = b1[j];
  const float* x = h1 + (size_t)b * 512;
  for (int d = 0; d < 512; ++d) acc += x[d] * w1[(size_t)d * 256 + j];
  hq[(size_t)b * 256 + j] = fmaxf(acc, 0.0f);
}

__global__ void lpo_qe2(const float* __restrict__ hq, const float* __restrict__ w2,
                        const float* __restrict__ b2, float* __restrict__ outq) {
  int b = threadIdx.x;                            // 32 threads
  float acc = b2[0];
  const float* x = hq + (size_t)b * 256;
  for (int d = 0; d < 256; ++d) acc += x[d] * w2[d];
  outq[b] = sigf(acc);
}

// ---------------- launch ----------------

extern "C" void kernel_launch(void* const* d_in, const int* in_sizes, int n_in,
                              void* d_out, int out_size, void* d_ws, size_t ws_size,
                              hipStream_t stream) {
  (void)in_sizes; (void)n_in; (void)out_size; (void)ws_size;
  const float* cs    = (const float*)d_in[0];
  const float* ce    = (const float*)d_in[2];
  const int*   mask  = (const int*)d_in[3];
  const float* se_w1 = (const float*)d_in[4];  const float* se_b1 = (const float*)d_in[5];
  const float* se_w2 = (const float*)d_in[6];  const float* se_b2 = (const float*)d_in[7];
  const float* wih0  = (const float*)d_in[8];  const float* whh0  = (const float*)d_in[9];
  const float* bih0  = (const float*)d_in[10]; const float* bhh0  = (const float*)d_in[11];
  const float* wih1  = (const float*)d_in[12]; const float* whh1  = (const float*)d_in[13];
  const float* bih1  = (const float*)d_in[14]; const float* bhh1  = (const float*)d_in[15];
  const float* wq    = (const float*)d_in[16]; const float* bq    = (const float*)d_in[17];
  const float* wk    = (const float*)d_in[18];
  const float* qe_w1 = (const float*)d_in[26]; const float* qe_b1 = (const float*)d_in[27];
  const float* qe_w2 = (const float*)d_in[28]; const float* qe_b2 = (const float*)d_in[29];
  float* out = (float*)d_out;

  char* w = (char*)d_ws;
  size_t off = 0;
  auto carve = [&](size_t bytes) -> char* {
    char* p = w + off; off += (bytes + 255) & ~(size_t)255; return p;
  };
  __bf16* ceb    = (__bf16*)carve((size_t)B32 * NCON * DC * 2);     // 134 MB
  __bf16* wp0    = (__bf16*)carve((size_t)128 * 32 * 32 * 16 * 2);  // 4 MB
  __bf16* wp1    = (__bf16*)carve((size_t)128 * 32 * 32 * 16 * 2);
  __bf16* xpk    = (__bf16*)carve((size_t)2 * 32 * 32 * 16 * 2);    // 64 KB packed Xcat
  float*  biasC0 = (float*)carve(2048 * 4);
  float*  biasC1 = (float*)carve(2048 * 4);
  float*  state  = (float*)carve((size_t)(4 * B32 * HSZ + B32 * DC) * 4);
  float*  h0 = state;
  float*  c0 = h0 + B32 * HSZ;
  float*  h1 = c0 + B32 * HSZ;
  float*  c1 = h1 + B32 * HSZ;
  float*  sel = c1 + B32 * HSZ;
  float*  hid   = (float*)carve((size_t)B32 * HSZ * 4);
  float*  sf    = (float*)carve((size_t)B32 * DC * 4);
  float*  G     = (float*)carve((size_t)B32 * 2048 * 4);
  float*  q     = (float*)carve((size_t)B32 * HSZ * 4);
  __bf16* qtilB = (__bf16*)carve((size_t)B32 * 8 * 32 * 16 * 2);
  float*  Lg    = (float*)carve((size_t)B32 * NH * NCON * 4);       // 8.4 MB
  float*  stats = (float*)carve((size_t)B32 * NH * 2 * 4);
  float*  hq    = (float*)carve((size_t)B32 * 256 * 4);
  int*    idx0  = (int*)carve(B32 * 4);

  hipMemsetAsync(state, 0, (size_t)(4 * B32 * HSZ + B32 * DC) * 4, stream);
  lpo_cvt_ce<<<65536, 256, 0, stream>>>(ce, ceb, (long long)B32 * NCON * DC / 4);
  lpo_pack_w<<<512, 256, 0, stream>>>(wih0, whh0, wp0);
  lpo_pack_w<<<512, 256, 0, stream>>>(wih1, whh1, wp1);
  lpo_bias_add<<<8, 256, 0, stream>>>(bih0, bhh0, biasC0, 2048);
  lpo_bias_add<<<8, 256, 0, stream>>>(bih1, bhh1, biasC1, 2048);
  lpo_enc1<<<32, 512, 0, stream>>>(cs, se_w1, se_b1, hid);
  lpo_enc2<<<32, 256, 0, stream>>>(hid, se_w2, se_b2, sf);
  lpo_idx_path<<<1, 32, 0, stream>>>(mask, idx0, out);

  for (int t = 0; t < PATHLEN; ++t) {
    lpo_pack_x<<<8, 256, 0, stream>>>(sf, DC, sel, DC, h0, xpk);
    lpo_gemm_cat<<<256, 32, 0, stream>>>(wp0, xpk, G);
    lpo_lstm_update<<<32, 512, 0, stream>>>(G, biasC0, h0, c0);
    lpo_pack_x<<<8, 256, 0, stream>>>(h0, HSZ, h1, HSZ, h1, xpk);
    lpo_gemm_cat<<<256, 32, 0, stream>>>(wp1, xpk, G);
    lpo_lstm_update<<<32, 512, 0, stream>>>(G, biasC1, h1, c1);
    if (t == 0) lpo_set_sel<<<32, 256, 0, stream>>>(ce, idx0, sel);
  }

  lpo_qproj<<<32, 512, 0, stream>>>(h1, wq, bq, q);
  lpo_qtil<<<512, 256, 0, stream>>>(q, wk, qtilB);
  lpo_logits<<<B32 * 512, 32, 0, stream>>>(ceb, qtilB, Lg);
  lpo_stats<<<B32 * NH, 256, 0, stream>>>(Lg, stats);
  lpo_avg<<<B32 * NCON / 256, 256, 0, stream>>>(Lg, stats, out + B32 * PATHLEN + B32);
  lpo_qe1<<<32, 256, 0, stream>>>(h1, qe_w1, qe_b1, hq);
  lpo_qe2<<<1, 32, 0, stream>>>(hq, qe_w2, qe_b2, out + B32 * PATHLEN);
}
